// DSS_67834713473246
// MI455X (gfx1250) — matrix operand order repaired and checked by reading, not compiled, not run on hardware
//
#include <hip/hip_runtime.h>
#include <hip/hip_bf16.h>
#include <math.h>

// ---------------------------------------------------------------------------
// DSS layer for MI455X (gfx1250), chunked modal-scan formulation.
//   y[b,h,l] = D[h]*u[b,h,l] + sum_{m<=l} K[h][l-m] * u[b,h,m]
//   K[h][d]  = Re( sum_n c_n r_n^d ),  r_n = exp(step*Lambda_n)
//   c_n      = (W_n/Lambda_n) * conj(s_n)/(|s_n|^2+EPS),  s_n = (1-r^L)/(1-r)
// Split: intra (lags 0..31) via WMMA f16 Toeplitz tiles; long range via
// N=64 complex modal states scanned chunk-by-chunk in fp32 VALU, with the
// per-chunk u staging done by the gfx1250 async copy unit (ASYNCcnt).
// ---------------------------------------------------------------------------

typedef __attribute__((ext_vector_type(16))) _Float16 v16h;
typedef __attribute__((ext_vector_type(8)))  float    v8f;

#define Bsz 8
#define Hsz 512
#define Nsz 64
#define Lsz 16384
#define CH  16
#define NCH (Lsz / CH)     // 1024 chunks
#define TAPS 32

// workspace layout (bytes); total ~8.75 MB
static constexpr size_t WS_A  = 0;                                  // f16  [Hsz][16][32]
static constexpr size_t WS_P  = WS_A  + (size_t)Hsz * 512  * 2;     // f32  [Hsz][16][128]
static constexpr size_t WS_Q  = WS_P  + (size_t)Hsz * 2048 * 4;     // f32  [Hsz][128][16]
static constexpr size_t WS_DC = WS_Q  + (size_t)Hsz * 2048 * 4;     // f32  [Hsz][128]

// ------------------------------ kernel 1: per-h setup -----------------------
// 512 blocks x 64 threads (thread = mode n). Deterministic LDS reduction for
// the Toeplitz taps (no fp atomics -> bitwise-stable across graph replays).
__global__ void dss_setup(const float* __restrict__ W,
                          const float* __restrict__ Lam,
                          const float* __restrict__ log_step,
                          _Float16* __restrict__ Aws,
                          float* __restrict__ Pws,
                          float* __restrict__ Qws,
                          float* __restrict__ DCws) {
  __shared__ float kpart[TAPS * Nsz];
  __shared__ float Ktap[TAPS];
  const int h = blockIdx.x;
  const int n = threadIdx.x;          // 0..63

  const float  PIF = 3.14159265358979323846f;
  const double PID = 3.141592653589793238462643383279502884;

  const float step = expf(log_step[h]);
  const float a  = -0.5f * step;
  const float b  = PIF * (float)n * step;
  const float ea = expf(a);
  const float cb = cosf(b), sb = sinf(b);
  const float rRe = ea * cb, rIm = ea * sb;

  // 1 - r, cancellation-free: 1 - e^a cos b = 2 sin^2(b/2) - cos(b)*expm1(a)
  const float sh2 = sinf(0.5f * b);
  const float dRe = 2.0f * sh2 * sh2 - cb * expm1f(a);
  const float dIm = -ea * sb;

  // r^L (tiny magnitude; double only for angle reduction)
  const float  eaL  = expf(a * (float)Lsz);
  const double angL = PID * (double)n * (double)step * (double)Lsz;
  const float  redL = (float)fmod(angL, 2.0 * PID);
  const float rLRe = eaL * cosf(redL);
  const float rLIm = eaL * sinf(redL);

  // s = (1 - r^L) / (1 - r)
  const float numRe = 1.0f - rLRe, numIm = -rLIm;
  const float den2  = dRe * dRe + dIm * dIm;
  const float sRe = (numRe * dRe + numIm * dIm) / den2;
  const float sIm = (numIm * dRe - numRe * dIm) / den2;

  // w / lambda
  const float lRe = Lam[(h * Nsz + n) * 2 + 0];
  const float lIm = Lam[(h * Nsz + n) * 2 + 1];
  const float wRe = W[(h * Nsz + n) * 2 + 0];
  const float wIm = W[(h * Nsz + n) * 2 + 1];
  const float l2  = lRe * lRe + lIm * lIm;
  const float wlRe = (wRe * lRe + wIm * lIm) / l2;
  const float wlIm = (wIm * lRe - wRe * lIm) / l2;

  // c = (w/lambda) * conj(s) / (|s|^2 + EPS)
  const float m2  = sRe * sRe + sIm * sIm + 1e-7f;
  const float cRe = (wlRe * sRe + wlIm * sIm) / m2;
  const float cIm = (wlIm * sRe - wlRe * sIm) / m2;

  // walk powers of r; emit taps (K), P (r^{16+i}), Q (r^{16-j}), decay (r^16)
  float pRe = 1.0f, pIm = 0.0f;        // r^i
  float cpRe = cRe, cpIm = cIm;        // c * r^i
  for (int i = 0; i < TAPS; ++i) {
    kpart[i * Nsz + n] = cpRe;
    if (i >= 16) {
      const int ii = i - 16;
      Pws[h * 2048 + ii * 128 + 2 * n + 0] =  cpRe;
      Pws[h * 2048 + ii * 128 + 2 * n + 1] = -cpIm;
    }
    if (i >= 1 && i <= 16) {
      const int j = 16 - i;
      Qws[h * 2048 + (2 * n + 0) * 16 + j] = pRe;
      Qws[h * 2048 + (2 * n + 1) * 16 + j] = pIm;
    }
    if (i == 16) {
      DCws[h * 128 + 2 * n + 0] = pRe;
      DCws[h * 128 + 2 * n + 1] = pIm;
    }
    float t;
    t = pRe * rRe - pIm * rIm;  pIm = pRe * rIm + pIm * rRe;  pRe = t;
    t = cpRe * rRe - cpIm * rIm; cpIm = cpRe * rIm + cpIm * rRe; cpRe = t;
  }
  __syncthreads();
  if (n < TAPS) {                       // deterministic tap reduction over modes
    float s = 0.0f;
    for (int k = 0; k < Nsz; ++k) s += kpart[n * Nsz + k];
    Ktap[n] = s;
  }
  __syncthreads();
  if (n < 16) {                         // Toeplitz band A[i][j] = K[i+16-j], f16
    for (int j = 0; j < 32; ++j) {
      const int d = n + 16 - j;
      const float v = (d >= 0 && d < TAPS) ? Ktap[d] : 0.0f;
      Aws[h * 512 + n * 32 + j] = (_Float16)v;
    }
  }
}

// --------------------- kernel 2: intra-chunk conv via WMMA ------------------
// grid = B*H blocks x 256 threads (8 waves). Each wave owns 8 output tiles
// of 16 positions x 16 chunks; one v_wmma_f32_16x16x32_f16 per tile.
// VGPR layouts per CDNA5 ISA:
//   A 16x32 f16: lane m=lane&15, kbase=(lane>>4)*8; elems 0..7 = K kbase..kbase+7,
//                elems 8..15 = K 16+kbase..16+kbase+7.
//   B 32x16 f16: lane col n=lane&15, kbase=(lane>>4)*8; same K order as A.
//   D 16x16 f32: lane col n=lane&15, rows m=(lane>>4)*8 + v.
__global__ void dss_intra(const float* __restrict__ u,
                          const _Float16* __restrict__ Aws,
                          float* __restrict__ out) {
  const int bh = blockIdx.x;
  const int b  = bh >> 9;             // /Hsz
  const int h  = bh & (Hsz - 1);
  const int lane = threadIdx.x & 31;
  const int wave = threadIdx.x >> 5;  // 0..7

  const float* ub = u   + ((size_t)(b * Hsz + h)) * Lsz;
  float*       yb = out + ((size_t)(b * Hsz + h)) * Lsz;

  // A operand: constant per h, load once
  const _Float16* Ah = Aws + h * 512;
  const int m  = lane & 15;
  const int kb = (lane >> 4) * 8;
  v16h av;
  for (int e = 0; e < 8; ++e) av[e]     = Ah[m * 32 + kb + e];
  for (int e = 0; e < 8; ++e) av[8 + e] = Ah[m * 32 + 16 + kb + e];

  const int ncol = lane & 15;
  const int half = lane >> 4;         // 0 or 1

  for (int it = 0; it < 8; ++it) {
    const int tile = wave * 8 + it;   // 64 tiles of 16 chunks
    const int q0   = tile * 16;
    // B operand: u window spanning chunks [q0+ncol-1, q0+ncol]
    const int chunk = q0 + ncol - 1 + half;
    v16h bv;
    for (int e = 0; e < 16; ++e) {
      const int kk = (e < 8) ? (half * 8 + e) : (16 + half * 8 + e - 8);
      const int c  = q0 + ncol - 1 + (kk >> 4);
      bv[e] = (c < 0) ? (_Float16)0.0f : (_Float16)ub[c * 16 + (kk & 15)];
    }
    // prefetch next tile's u window (global_prefetch_b8)
    {
      int pc = chunk + 16; if (pc > NCH - 1) pc = NCH - 1; if (pc < 0) pc = 0;
      __builtin_prefetch(ub + pc * 16, 0, 1);
    }
    v8f cv = {};
    cv = __builtin_amdgcn_wmma_f32_16x16x32_f16(
        /*neg_a=*/false, av, /*neg_b=*/false, bv,
        /*c_mod=*/(short)0, cv, /*reuse_a=*/false, /*reuse_b=*/false);
    float* yp = yb + (q0 + ncol) * 16 + half * 8;
    for (int e = 0; e < 8; ++e) yp[e] = cv[e];   // overwrite (d_out was poisoned)
  }
}

// --------------------- kernel 3: modal state scan (long range) --------------
// grid = H blocks x 256 threads; each block handles all B=8 batches of its h.
// Per-h matrices staged once in LDS; 1024 sequential chunk steps. The u chunk
// (8 batches x 16 taps = 512 B) is staged each step by wave 4 through the
// gfx1250 async copy unit: 32 lanes x global_load_async_to_lds_b128, then
// s_wait_asynccnt 0 + workgroup barrier for cross-wave visibility.
__global__ void dss_scan(const float* __restrict__ u,
                         const float* __restrict__ Dvec,
                         const float* __restrict__ Pws,
                         const float* __restrict__ Qws,
                         const float* __restrict__ DCws,
                         float* __restrict__ out) {
  __shared__ float lP[2048];                       // [16][128]
  __shared__ float lQ[2048];                       // [128][16]
  __shared__ float lDC[128];
  __shared__ float Sv[Bsz * 128];                  // complex states, Re/Im interleaved
  __shared__ __align__(16) float uS[2][Bsz * 16];  // double-buffered u chunks
  const int h = blockIdx.x;
  const int t = threadIdx.x;            // 0..255

  for (int k = t; k < 2048; k += 256) { lP[k] = Pws[h * 2048 + k]; lQ[k] = Qws[h * 2048 + k]; }
  if (t < 128) lDC[t] = DCws[h * 128 + t];
  for (int k = t; k < Bsz * 128; k += 256) Sv[k] = 0.0f;
  const float Dh = Dvec[h];
  __syncthreads();

  for (int q = 0; q < NCH; ++q) {
    // ---- stage u chunk q via async copy (wave 4: lanes cover 8 b x 16 j) ----
    if (t >= 128 && t < 160) {
      const int lane = t - 128;                    // 0..31, 4 floats each
      const int b  = lane >> 2;                    // batch
      const int j4 = (lane & 3) * 4;               // tap group
      const float* gp = u + ((size_t)(b * Hsz + h)) * Lsz + q * 16 + j4;
      const unsigned long long ga = (unsigned long long)gp;
      const unsigned lds = (unsigned)(unsigned long long)&uS[q & 1][lane * 4];
      asm volatile("global_load_async_to_lds_b128 %0, %1, off"
                   :: "v"(lds), "v"(ga) : "memory");
    }
    asm volatile("s_wait_asynccnt 0" ::: "memory"); // issuing wave drains; others no-op
    __syncthreads();
    // ---- inter-chunk output: y[b, q*16+i] += D*u + Re(P_i . S_b) ----
    if (t < 128) {
      const int b = t >> 4, i = t & 15;
      float acc = Dh * uS[q & 1][t];
      const float* Pi = &lP[i * 128];
      const float* Sb = &Sv[b * 128];
      for (int k = 0; k < 128; ++k) acc += Pi[k] * Sb[k];
      out[((size_t)(b * Hsz + h)) * Lsz + q * 16 + i] += acc;
    }
    __syncthreads();
    // ---- state update: S(q) = r^16 * S(q-1) + Q . u[chunk q-1] ----
    if (q >= 1) {
      const float* up = uS[(q - 1) & 1];
      for (int idx = t; idx < Bsz * Nsz; idx += 256) {
        const int b = idx >> 6, n = idx & 63;
        const float re = Sv[b * 128 + 2 * n + 0];
        const float im = Sv[b * 128 + 2 * n + 1];
        const float dr = lDC[2 * n + 0], di = lDC[2 * n + 1];
        float nr = dr * re - di * im;
        float ni = dr * im + di * re;
        const float* qr = &lQ[(2 * n + 0) * 16];
        const float* qi = &lQ[(2 * n + 1) * 16];
        const float* ux = &up[b * 16];
        for (int j = 0; j < 16; ++j) { nr += qr[j] * ux[j]; ni += qi[j] * ux[j]; }
        Sv[b * 128 + 2 * n + 0] = nr;
        Sv[b * 128 + 2 * n + 1] = ni;
      }
    }
    // next iteration's post-stage __syncthreads orders Sv writes before reads
  }
}

// ------------------------------ launcher ------------------------------------
extern "C" void kernel_launch(void* const* d_in, const int* in_sizes, int n_in,
                              void* d_out, int out_size, void* d_ws, size_t ws_size,
                              hipStream_t stream) {
  (void)in_sizes; (void)n_in; (void)out_size; (void)ws_size;
  const float* u     = (const float*)d_in[0];   // (B,H,L)
  const float* W     = (const float*)d_in[1];   // (H,N,2)
  const float* Lam   = (const float*)d_in[2];   // (H,N,2)
  const float* lstep = (const float*)d_in[3];   // (H,)
  const float* Dv    = (const float*)d_in[4];   // (H,)
  float* out = (float*)d_out;

  char* ws = (char*)d_ws;                       // needs ~8.75 MB
  _Float16* Aws = (_Float16*)(ws + WS_A);
  float*    Pws = (float*)(ws + WS_P);
  float*    Qws = (float*)(ws + WS_Q);
  float*    DCws = (float*)(ws + WS_DC);

  dss_setup<<<Hsz, Nsz, 0, stream>>>(W, Lam, lstep, Aws, Pws, Qws, DCws);
  dss_intra<<<Bsz * Hsz, 256, 0, stream>>>(u, Aws, out);
  dss_scan<<<Hsz, 256, 0, stream>>>(u, Dv, Pws, Qws, DCws, out);
}
